// ViTC_64450279244197
// MI455X (gfx1250) — compile-verified, matrix operand order checked by reading, not run on hardware
//
#include <hip/hip_runtime.h>
#include <hip/hip_bf16.h>

// ---------------------------------------------------------------------------
// ViT-C forward for MI455X (gfx1250): fp16 WMMA GEMMs (fp32 accum) for all
// large matmuls, fused WMMA attention, fp32 direct conv stem.
// Round 3: B tile and V tile stored TRANSPOSED in LDS so all WMMA fragment
// reads (A and B) are contiguous ds_load_b128, eliminating the ds_load_u16 +
// v_mov_b16 merge storm in the inner loops.
// ---------------------------------------------------------------------------

typedef __attribute__((ext_vector_type(16))) _Float16 v16h;
typedef __attribute__((ext_vector_type(4)))  _Float16 v4h;
typedef __attribute__((ext_vector_type(8)))  float    v8f;

#define HEADS 12
#define DH    64
#define EMB   768
#define SEQ   197
#define SEQP  224            // keys padded to 7 * 32
#define QKVLD 2304
#define NTOK  (32 * SEQ)     // 6304 rows for token GEMMs

// K index held by this lane for half j of a 16x32 f16 A/B fragment
// (CDNA5 ISA 7.12.2: lanes 0-15 hold K {0..7,16..23}, lanes 16-31 {8..15,24..31})
__device__ __forceinline__ int wmma_k(int lane, int j) {
    int kb = (lane >> 4) << 3;
    return (j < 8) ? (kb + j) : (16 + kb + (j - 8));
}

// ---------------------------------------------------------------------------
// Direct conv + BN + ReLU (fp32) — conv0..conv3 (~4% of total FLOPs)
// ---------------------------------------------------------------------------
__global__ void conv_bn_relu_kernel(const float* __restrict__ in, const float* __restrict__ w,
                                    const float* __restrict__ g, const float* __restrict__ bta,
                                    const float* __restrict__ mu, const float* __restrict__ var,
                                    float* __restrict__ out,
                                    int Cin, int H, int W, int Cout, int Ho, int Wo,
                                    int ksz, int stride, int pad, int total) {
    int idx = blockIdx.x * blockDim.x + threadIdx.x;
    if (idx >= total) return;
    int wo = idx % Wo;
    int t  = idx / Wo;
    int ho = t % Ho;  t /= Ho;
    int co = t % Cout;
    int b  = t / Cout;
    float s = 0.f;
    const float* wp = w + (size_t)co * Cin * ksz * ksz;
    for (int ci = 0; ci < Cin; ++ci) {
        const float* ip = in + ((size_t)(b * Cin + ci) * H) * W;
        for (int ky = 0; ky < ksz; ++ky) {
            int iy = ho * stride + ky - pad;
            if ((unsigned)iy >= (unsigned)H) continue;
            for (int kx = 0; kx < ksz; ++kx) {
                int ix = wo * stride + kx - pad;
                if ((unsigned)ix >= (unsigned)W) continue;
                s += ip[iy * W + ix] * wp[(ci * ksz + ky) * ksz + kx];
            }
        }
    }
    float y = g[co] * (s - mu[co]) * rsqrtf(var[co] + 1e-5f) + bta[co];
    out[idx] = fmaxf(y, 0.f);
}

// ---------------------------------------------------------------------------
// WMMA GEMM: C[M,N] = act(A[M,K] * B[K,N] + bias + resid)
// Block tile 128x64, 8 waves, wave tile 32x32 (2x2 v_wmma_f32_16x16x32_f16).
// Register double-buffered staging; A stored [row][k], B stored TRANSPOSED
// [col][k] so both fragment types read contiguous 16-byte LDS chunks.
// ---------------------------------------------------------------------------
#define BM 128
#define BN 64
#define BK 32

__global__ __launch_bounds__(256) void gemm_bias_act(
        const float* __restrict__ A, int lda,
        const float* __restrict__ Bm, int ldb,
        float* __restrict__ C, int ldc,
        const float* __restrict__ bias,
        const float* __restrict__ resid,
        int M, int N, int K, int relu) {
    __shared__ _Float16 As[BM][BK + 8];
    __shared__ _Float16 Bs[BN][BK + 8];   // transposed: [n][k]

    const int tid  = threadIdx.x;
    const int lane = tid & 31;
    const int wave = tid >> 5;
    const int wm   = (wave & 3) << 5;   // 0,32,64,96
    const int wn   = (wave >> 2) << 5;  // 0,32
    const int bm0  = blockIdx.y * BM;
    const int bn0  = blockIdx.x * BN;
    const int lrow = lane & 15;
    const bool interior = (bm0 + BM <= M) && (bn0 + BN <= N);

    // per-thread staging registers: A 4x float4 (row-major), B 2x float4
    // (B group g: column bn0+(G&63), rows k0+(G>>6)*4 .. +3, G = tid + g*256)
    float4 ra[4], rb[2];

    auto load_tiles = [&](int k0) {
        if (interior && (k0 + BK) <= K) {
#pragma unroll
            for (int i = 0; i < 4; ++i) {
                int l = tid * 16 + i * 4;
                int r = l >> 5, c = l & 31;
                ra[i] = *(const float4*)&A[(size_t)(bm0 + r) * lda + k0 + c];
            }
#pragma unroll
            for (int g = 0; g < 2; ++g) {
                int G = tid + g * 256;
                int col = G & 63, r0 = (G >> 6) * 4;
                const float* bp = Bm + (size_t)(k0 + r0) * ldb + bn0 + col;
                rb[g] = make_float4(bp[0], bp[ldb], bp[2 * (size_t)ldb], bp[3 * (size_t)ldb]);
            }
        } else {
            // branch-free edge path: clamp address, select after load
#pragma unroll
            for (int i = 0; i < 4; ++i) {
                int l = tid * 16 + i * 4;
                int r = l >> 5, c = l & 31;
                float tmp[4];
#pragma unroll
                for (int u = 0; u < 4; ++u) {
                    int gr = bm0 + r, gc = k0 + c + u;
                    int grc = (gr < M) ? gr : 0;
                    int gcc = (gc < K) ? gc : 0;
                    float v = A[(size_t)grc * lda + gcc];
                    tmp[u] = (gr < M && gc < K) ? v : 0.f;
                }
                ra[i] = make_float4(tmp[0], tmp[1], tmp[2], tmp[3]);
            }
#pragma unroll
            for (int g = 0; g < 2; ++g) {
                int G = tid + g * 256;
                int col = G & 63, r0 = (G >> 6) * 4;
                float tmp[4];
#pragma unroll
                for (int u = 0; u < 4; ++u) {
                    int gr = k0 + r0 + u, gc = bn0 + col;
                    int grc = (gr < K) ? gr : 0;
                    int gcc = (gc < N) ? gc : 0;
                    float v = Bm[(size_t)grc * ldb + gcc];
                    tmp[u] = (gr < K && gc < N) ? v : 0.f;
                }
                rb[g] = make_float4(tmp[0], tmp[1], tmp[2], tmp[3]);
            }
        }
    };

    auto store_tiles = [&]() {
#pragma unroll
        for (int i = 0; i < 4; ++i) {
            int l = tid * 16 + i * 4;
            int r = l >> 5, c = l & 31;
            v4h hv;
            hv[0] = (_Float16)ra[i].x; hv[1] = (_Float16)ra[i].y;
            hv[2] = (_Float16)ra[i].z; hv[3] = (_Float16)ra[i].w;
            *(v4h*)&As[r][c] = hv;
        }
#pragma unroll
        for (int g = 0; g < 2; ++g) {
            int G = tid + g * 256;
            int col = G & 63, r0 = (G >> 6) * 4;
            v4h hv;
            hv[0] = (_Float16)rb[g].x; hv[1] = (_Float16)rb[g].y;
            hv[2] = (_Float16)rb[g].z; hv[3] = (_Float16)rb[g].w;
            *(v4h*)&Bs[col][r0] = hv;
        }
    };

    v8f acc[2][2] = {};

    load_tiles(0);
    const int nk = (K + BK - 1) / BK;
    for (int t = 0; t < nk; ++t) {
        store_tiles();
        __syncthreads();
        if (t + 1 < nk) load_tiles((t + 1) * BK);   // issue next loads early

        v16h af[2], bf[2];
#pragma unroll
        for (int i = 0; i < 2; ++i)
#pragma unroll
            for (int j = 0; j < 16; ++j) {
                int kk = wmma_k(lane, j);
                af[i][j] = As[wm + i * 16 + lrow][kk];
                bf[i][j] = Bs[wn + i * 16 + lrow][kk];
            }
#pragma unroll
        for (int i = 0; i < 2; ++i)
#pragma unroll
            for (int j = 0; j < 2; ++j)
                acc[i][j] = __builtin_amdgcn_wmma_f32_16x16x32_f16(
                    false, af[i], false, bf[j], (short)0, acc[i][j], false, false);
        __syncthreads();
    }

    // epilogue: C/D layout — VGPR r holds (M = r + 8*(lane>>4), N = lane&15)
    if (interior) {
#pragma unroll
        for (int i = 0; i < 2; ++i)
#pragma unroll
            for (int j = 0; j < 2; ++j) {
                int gc = bn0 + wn + j * 16 + lrow;
                float bval = bias ? bias[gc] : 0.f;
#pragma unroll
                for (int r = 0; r < 8; ++r) {
                    int gr = bm0 + wm + i * 16 + r + ((lane >> 4) << 3);
                    float v = acc[i][j][r] + bval;
                    if (resid) v += resid[(size_t)gr * ldc + gc];
                    if (relu)  v = fmaxf(v, 0.f);
                    C[(size_t)gr * ldc + gc] = v;
                }
            }
    } else {
#pragma unroll
        for (int i = 0; i < 2; ++i)
#pragma unroll
            for (int j = 0; j < 2; ++j) {
                int gc = bn0 + wn + j * 16 + lrow;
                if (gc >= N) continue;
                float bval = bias ? bias[gc] : 0.f;
#pragma unroll
                for (int r = 0; r < 8; ++r) {
                    int gr = bm0 + wm + i * 16 + r + ((lane >> 4) << 3);
                    if (gr >= M) continue;
                    float v = acc[i][j][r] + bval;
                    if (resid) v += resid[(size_t)gr * ldc + gc];
                    if (relu)  v = fmaxf(v, 0.f);
                    C[(size_t)gr * ldc + gc] = v;
                }
            }
    }
}

// ---------------------------------------------------------------------------
// Fused attention per (b,h): WMMA Q*K^T -> softmax (LDS) -> WMMA P*V
// Reference divides logits by dh^-0.5, i.e. multiplies by sqrt(64)=8.
// Block: 64 threads = 2 waves, each wave owns 16 query rows.
// V is staged TRANSPOSED into LDS ([d][key], f16, keys>=197 zeroed), unioned
// with the logits buffer (dead after softmax): max(28672, 29696) + 14336 B.
// ---------------------------------------------------------------------------
#define VTLD 232   // padded transposed-V row stride (halves): SEQP + 8

__global__ __launch_bounds__(64) void attention_kernel(const float* __restrict__ QKV,
                                                       float* __restrict__ Out) {
    // union: phase 1-2 -> float logits [2][16][SEQP]   (28672 B)
    //        phase 3   -> f16 V^T tile [DH][VTLD]      (29696 B)
    __shared__ __align__(16) unsigned char uS[DH * VTLD * 2];
    float*    sS  = reinterpret_cast<float*>(uS);      // [(wave*16+row)*SEQP + col]
    _Float16* sVt = reinterpret_cast<_Float16*>(uS);   // [d*VTLD + key]
    __shared__ _Float16 sP[2][16][SEQP];

    const int tid  = threadIdx.x;
    const int wave = tid >> 5;
    const int lane = tid & 31;
    const int b    = blockIdx.x / HEADS;
    const int h    = blockIdx.x % HEADS;
    const int q0   = blockIdx.y * 32 + wave * 16;
    const int lrow = lane & 15;
    const int kb   = (lane >> 4) << 3;

    // ---- Q fragments: 16x64 -> two 16x32 f16 fragments (float4 loads) ----
    v16h aq[2];
    {
        int  qr  = q0 + lrow;
        bool okq = qr < SEQ;
        const float* qp = QKV + (size_t)(b * SEQ + (okq ? qr : 0)) * QKVLD + h * DH;
#pragma unroll
        for (int s = 0; s < 2; ++s) {
            int d0 = s * 32 + kb;
            float4 f0 = *(const float4*)(qp + d0);
            float4 f1 = *(const float4*)(qp + d0 + 4);
            float4 f2 = *(const float4*)(qp + d0 + 16);
            float4 f3 = *(const float4*)(qp + d0 + 20);
            float tf[16] = { f0.x, f0.y, f0.z, f0.w, f1.x, f1.y, f1.z, f1.w,
                             f2.x, f2.y, f2.z, f2.w, f3.x, f3.y, f3.z, f3.w };
#pragma unroll
            for (int j = 0; j < 16; ++j)
                aq[s][j] = okq ? (_Float16)tf[j] : (_Float16)0.f;
        }
    }

    // ---- logits: 14 key tiles of 16 ----
    for (int jt = 0; jt < SEQP / 16; ++jt) {
        v8f acc = {};
        int  key = jt * 16 + lrow;
        bool okk = key < SEQ;
        const float* kp = QKV + (size_t)(b * SEQ + (okk ? key : 0)) * QKVLD + EMB + h * DH;
#pragma unroll
        for (int s = 0; s < 2; ++s) {
            int d0 = s * 32 + kb;
            float4 f0 = *(const float4*)(kp + d0);
            float4 f1 = *(const float4*)(kp + d0 + 4);
            float4 f2 = *(const float4*)(kp + d0 + 16);
            float4 f3 = *(const float4*)(kp + d0 + 20);
            float tf[16] = { f0.x, f0.y, f0.z, f0.w, f1.x, f1.y, f1.z, f1.w,
                             f2.x, f2.y, f2.z, f2.w, f3.x, f3.y, f3.z, f3.w };
            v16h bk;
#pragma unroll
            for (int j = 0; j < 16; ++j)
                bk[j] = okk ? (_Float16)tf[j] : (_Float16)0.f;
            acc = __builtin_amdgcn_wmma_f32_16x16x32_f16(
                false, aq[s], false, bk, (short)0, acc, false, false);
        }
        int col = jt * 16 + lrow;
#pragma unroll
        for (int r = 0; r < 8; ++r) {
            int row = r + ((lane >> 4) << 3);
            sS[(wave * 16 + row) * SEQP + col] = (col < SEQ) ? acc[r] * 8.0f : -1e30f;
        }
    }
    __syncthreads();

    // ---- softmax: 2 lanes per row (cols split 112/112), combine via shfl ----
    {
        int row  = lane & 15;
        int half = lane >> 4;
        int c0   = half * (SEQP / 2);
        float* sr = &sS[(wave * 16 + row) * SEQP];
        float mx = -1e30f;
        for (int c = c0; c < c0 + SEQP / 2; ++c) mx = fmaxf(mx, sr[c]);
        mx = fmaxf(mx, __shfl_xor(mx, 16, 32));
        float sum = 0.f;
        for (int c = c0; c < c0 + SEQP / 2; ++c) {
            float e = __expf(sr[c] - mx);
            sr[c] = e;
            sum += e;
        }
        sum += __shfl_xor(sum, 16, 32);
        float inv = 1.0f / sum;
        for (int c = c0; c < c0 + SEQP / 2; ++c)
            sP[wave][row][c] = (_Float16)(sr[c] * inv);
    }
    __syncthreads();   // sS dead from here; reuse as sVt

    // ---- stage V^T tile [DH][SEQP] -> LDS f16 (keys >= SEQ zeroed) ----
    for (int l = tid; l < (SEQP * DH) / 4; l += 64) {
        int e   = l * 4;
        int key = e >> 6;        // / DH
        int d0  = e & 63;
        const float* vp = QKV + (size_t)(b * SEQ + (key < SEQ ? key : 0)) * QKVLD
                        + 2 * EMB + h * DH + d0;
        float4 f = *(const float4*)vp;
        bool ok = key < SEQ;
        sVt[(d0 + 0) * VTLD + key] = ok ? (_Float16)f.x : (_Float16)0.f;
        sVt[(d0 + 1) * VTLD + key] = ok ? (_Float16)f.y : (_Float16)0.f;
        sVt[(d0 + 2) * VTLD + key] = ok ? (_Float16)f.z : (_Float16)0.f;
        sVt[(d0 + 3) * VTLD + key] = ok ? (_Float16)f.w : (_Float16)0.f;
    }
    __syncthreads();

    // ---- out = P @ V : 16x64 = 4 column tiles, 7 K-steps of 32 ----
#pragma unroll
    for (int t = 0; t < 4; ++t) {
        v8f acc = {};
        for (int ks = 0; ks < SEQP / 32; ++ks) {
            v16h ap, bv;
#pragma unroll
            for (int j = 0; j < 16; ++j) {
                int kk = ks * 32 + wmma_k(lane, j);
                ap[j] = sP[wave][lrow][kk];
                bv[j] = sVt[(t * 16 + lrow) * VTLD + kk];
            }
            acc = __builtin_amdgcn_wmma_f32_16x16x32_f16(
                false, ap, false, bv, (short)0, acc, false, false);
        }
        int col = h * DH + t * 16 + lrow;
#pragma unroll
        for (int r = 0; r < 8; ++r) {
            int qr = q0 + r + ((lane >> 4) << 3);
            if (qr < SEQ)
                Out[(size_t)(b * SEQ + qr) * EMB + col] = acc[r];
        }
    }
}

// ---------------------------------------------------------------------------
// LayerNorm faithful to reference: unbiased std /(C-1), eps added to std
// ---------------------------------------------------------------------------
__global__ __launch_bounds__(256) void ln_kernel(const float* __restrict__ x, size_t rstride,
                                                 const float* __restrict__ a,
                                                 const float* __restrict__ b,
                                                 float* __restrict__ y, size_t ostride, int C) {
    __shared__ float red[256];
    __shared__ float stats[2];
    int row = blockIdx.x;
    int tid = threadIdx.x;
    const float* xr = x + (size_t)row * rstride;

    float s = 0.f;
    for (int c = tid; c < C; c += 256) s += xr[c];
    red[tid] = s; __syncthreads();
    for (int off = 128; off > 0; off >>= 1) { if (tid < off) red[tid] += red[tid + off]; __syncthreads(); }
    if (tid == 0) stats[0] = red[0] / (float)C;
    __syncthreads();
    float mu = stats[0];

    float s2 = 0.f;
    for (int c = tid; c < C; c += 256) { float d = xr[c] - mu; s2 += d * d; }
    red[tid] = s2; __syncthreads();
    for (int off = 128; off > 0; off >>= 1) { if (tid < off) red[tid] += red[tid + off]; __syncthreads(); }
    if (tid == 0) stats[1] = sqrtf(red[0] / (float)(C - 1));
    __syncthreads();

    float inv = 1.0f / (stats[1] + 1e-4f);
    for (int c = tid; c < C; c += 256)
        y[(size_t)row * ostride + c] = a[c] * (xr[c] - mu) * inv + b[c];
}

// ---------------------------------------------------------------------------
// conv4 prep: transpose [32,512,196] -> [6272,512] and weight [768,512]->[512,768]
// ---------------------------------------------------------------------------
__global__ void transpose_c3_kernel(const float* __restrict__ in, float* __restrict__ out, int total) {
    int idx = blockIdx.x * blockDim.x + threadIdx.x;
    if (idx >= total) return;
    int p  = idx % 196;
    int t  = idx / 196;
    int ci = t % 512;
    int b  = t / 512;
    out[(size_t)(b * 196 + p) * 512 + ci] = in[idx];
}

__global__ void transpose_w4_kernel(const float* __restrict__ w, float* __restrict__ out) {
    int idx = blockIdx.x * blockDim.x + threadIdx.x;
    if (idx >= 768 * 512) return;
    int k = idx % 512;
    int n = idx / 512;
    out[(size_t)k * 768 + n] = w[idx];
}

// tokens: BN+ReLU on conv4 raw output, cls token at t=0, add pos-enc
__global__ void build_tokens_kernel(const float* __restrict__ T4, const float* __restrict__ cls,
                                    const float* __restrict__ pos, const float* __restrict__ g,
                                    const float* __restrict__ bb, const float* __restrict__ mu,
                                    const float* __restrict__ var, float* __restrict__ X, int total) {
    int idx = blockIdx.x * blockDim.x + threadIdx.x;
    if (idx >= total) return;
    int c = idx % EMB;
    int t = (idx / EMB) % SEQ;
    int b = idx / (EMB * SEQ);
    float val;
    if (t == 0) {
        val = cls[c];
    } else {
        int m = b * 196 + (t - 1);
        float raw = T4[(size_t)m * EMB + c];
        float y = g[c] * (raw - mu[c]) * rsqrtf(var[c] + 1e-5f) + bb[c];
        val = fmaxf(y, 0.f);
    }
    X[idx] = val + pos[t * EMB + c];
}

// ---------------------------------------------------------------------------
// Host orchestration
// ---------------------------------------------------------------------------
extern "C" void kernel_launch(void* const* d_in, const int* in_sizes, int n_in,
                              void* d_out, int out_size, void* d_ws, size_t ws_size,
                              hipStream_t stream) {
    (void)in_sizes; (void)n_in; (void)out_size; (void)ws_size;

    const float* x_in = (const float*)d_in[0];
    const float *cw[5], *bg[5], *bb[5], *bm[5], *bv[5];
    for (int i = 0; i < 5; ++i) {
        cw[i] = (const float*)d_in[1 + 5 * i];
        bg[i] = (const float*)d_in[2 + 5 * i];
        bb[i] = (const float*)d_in[3 + 5 * i];
        bm[i] = (const float*)d_in[4 + 5 * i];
        bv[i] = (const float*)d_in[5 + 5 * i];
    }
    const float* cls    = (const float*)d_in[26];
    const float* pos    = (const float*)d_in[27];
    const float* ln1_a  = (const float*)d_in[28];
    const float* ln1_b  = (const float*)d_in[29];
    const float* qkv_w  = (const float*)d_in[30];
    const float* qkv_b  = (const float*)d_in[31];
    const float* proj_w = (const float*)d_in[32];
    const float* proj_b = (const float*)d_in[33];
    const float* ln2_a  = (const float*)d_in[34];
    const float* ln2_b  = (const float*)d_in[35];
    const float* ff1_w  = (const float*)d_in[36];
    const float* ff1_b  = (const float*)d_in[37];
    const float* ff2_w  = (const float*)d_in[38];
    const float* ff2_b  = (const float*)d_in[39];
    const float* norm_a = (const float*)d_in[40];
    const float* norm_b = (const float*)d_in[41];
    const float* fc_w   = (const float*)d_in[42];
    const float* fc_b   = (const float*)d_in[43];

    // workspace layout (floats)
    float* ws = (float*)d_ws;
    size_t off = 0;
    auto alloc = [&](size_t n) { float* p = ws + off; off += n; return p; };
    float* c0  = alloc((size_t)32 * 64 * 112 * 112);
    float* c1  = alloc((size_t)32 * 128 * 56 * 56);
    float* c2  = alloc((size_t)32 * 256 * 28 * 28);
    float* c3  = alloc((size_t)32 * 512 * 14 * 14);
    float* A4  = alloc((size_t)6272 * 512);
    float* T4  = alloc((size_t)6272 * 768);
    float* W4T = alloc((size_t)512 * 768);
    float* X   = alloc((size_t)NTOK * EMB);
    float* Y   = alloc((size_t)NTOK * EMB);
    float* QKV = alloc((size_t)NTOK * QKVLD);
    float* ATT = alloc((size_t)NTOK * EMB);
    float* X2  = alloc((size_t)NTOK * EMB);
    float* Hf  = alloc((size_t)NTOK * 3072);
    float* CLS = alloc((size_t)32 * EMB);

    auto gemm = [&](const float* A, int lda, const float* B, int ldb, float* C, int ldc,
                    const float* bias, const float* resid, int M, int N, int K, int relu) {
        dim3 g((N + BN - 1) / BN, (M + BM - 1) / BM);
        gemm_bias_act<<<g, 256, 0, stream>>>(A, lda, B, ldb, C, ldc, bias, resid, M, N, K, relu);
    };

    // ---- conv stem (fp32 direct) ----
    struct CD { const float* in; float* out; int Cin, H, W, Cout, Ho, Wo, k, s, p; };
    CD cds[4] = {
        { x_in, c0,   3, 224, 224,  64, 112, 112, 3, 2, 1 },
        { c0,   c1,  64, 112, 112, 128,  56,  56, 3, 2, 1 },
        { c1,   c2, 128,  56,  56, 256,  28,  28, 3, 2, 1 },
        { c2,   c3, 256,  28,  28, 512,  14,  14, 3, 2, 1 },
    };
    for (int i = 0; i < 4; ++i) {
        int total = 32 * cds[i].Cout * cds[i].Ho * cds[i].Wo;
        conv_bn_relu_kernel<<<(total + 255) / 256, 256, 0, stream>>>(
            cds[i].in, cw[i], bg[i], bb[i], bm[i], bv[i], cds[i].out,
            cds[i].Cin, cds[i].H, cds[i].W, cds[i].Cout, cds[i].Ho, cds[i].Wo,
            cds[i].k, cds[i].s, cds[i].p, total);
    }

    // ---- conv4 (1x1) as WMMA GEMM + token build ----
    {
        int tA = 32 * 512 * 196;
        transpose_c3_kernel<<<(tA + 255) / 256, 256, 0, stream>>>(c3, A4, tA);
        transpose_w4_kernel<<<(768 * 512 + 255) / 256, 256, 0, stream>>>(cw[4], W4T);
        gemm(A4, 512, W4T, 768, T4, 768, nullptr, nullptr, 6272, 768, 512, 0);
        int tX = 32 * SEQ * EMB;
        build_tokens_kernel<<<(tX + 255) / 256, 256, 0, stream>>>(
            T4, cls, pos, bg[4], bb[4], bm[4], bv[4], X, tX);
    }

    // ---- 12 transformer layers ----
    for (int l = 0; l < 12; ++l) {
        const float* l1a = ln1_a + (size_t)l * EMB;
        const float* l1b = ln1_b + (size_t)l * EMB;
        const float* qw  = qkv_w + (size_t)l * EMB * QKVLD;
        const float* qb  = qkv_b + (size_t)l * QKVLD;
        const float* pw  = proj_w + (size_t)l * EMB * EMB;
        const float* pb  = proj_b + (size_t)l * EMB;
        const float* l2a = ln2_a + (size_t)l * EMB;
        const float* l2b = ln2_b + (size_t)l * EMB;
        const float* f1w = ff1_w + (size_t)l * EMB * 3072;
        const float* f1b = ff1_b + (size_t)l * 3072;
        const float* f2w = ff2_w + (size_t)l * 3072 * EMB;
        const float* f2b = ff2_b + (size_t)l * EMB;

        // y = LN1(x)
        ln_kernel<<<NTOK, 256, 0, stream>>>(X, (size_t)EMB, l1a, l1b, Y, (size_t)EMB, EMB);
        // qkv = y @ qw + qb
        gemm(Y, EMB, qw, QKVLD, QKV, QKVLD, qb, nullptr, NTOK, QKVLD, EMB, 0);
        // fused attention -> ATT [NTOK, EMB]
        {
            dim3 g(32 * HEADS, SEQP / 32);
            attention_kernel<<<g, 64, 0, stream>>>(QKV, ATT);
        }
        // x2 = x + att @ pw + pb
        gemm(ATT, EMB, pw, EMB, X2, EMB, pb, X, NTOK, EMB, EMB, 0);
        // y = LN2(x2)
        ln_kernel<<<NTOK, 256, 0, stream>>>(X2, (size_t)EMB, l2a, l2b, Y, (size_t)EMB, EMB);
        // h = relu(y @ f1w + f1b)
        gemm(Y, EMB, f1w, 3072, Hf, 3072, f1b, nullptr, NTOK, 3072, EMB, 1);
        // x = x2 + h @ f2w + f2b
        gemm(Hf, 3072, f2w, EMB, X, EMB, f2b, X2, NTOK, EMB, 3072, 0);
    }

    // ---- final LN on cls rows + head ----
    ln_kernel<<<32, 256, 0, stream>>>(X, (size_t)SEQ * EMB, norm_a, norm_b, CLS, (size_t)EMB, EMB);
    gemm(CLS, EMB, fc_w, 1000, (float*)d_out, 1000, fc_b, nullptr, 32, 1000, EMB, 0);
}
